// Model_26233660244168
// MI455X (gfx1250) — matrix operand order repaired and checked by reading, not compiled, not run on hardware
//
#include <hip/hip_runtime.h>
#include <hip/hip_bf16.h>

#define B_SZ   4096
#define D_IN   512
#define D_H    1024
#define D_OUT  256
#define NSTEP  25
#define BETA   0.9f
#define THR    1.0f
#define DROP_SCALE 1.25f   // 1/(1-0.2)

typedef __attribute__((ext_vector_type(16))) _Float16 v16h;
typedef __attribute__((ext_vector_type(8)))  float    v8f;

union FragH { v16h v; uint4 q[2]; };

// A fragment: lane<16 -> row m, K {k..k+7, k+16..k+23}; lane>=16 shifts by 8.
__device__ __forceinline__ void load_fragA(FragH& f, const _Float16* p) {
    f.q[0] = *(const uint4*)(p);
    f.q[1] = *(const uint4*)(p + 16);
}
// B fragment: lane<16 -> col n, K {k..k+7, k+16..k+23}; lane>=16 shifts by 8.
__device__ __forceinline__ void load_fragB(FragH& f, const _Float16* p) {
    f.q[0] = *(const uint4*)(p);
    f.q[1] = *(const uint4*)(p + 16);
}

// Async global->LDS copy (CDNA5, tracked by ASYNCcnt). Immediate byte offset
// keeps the global base pointer loop-invariant.
__device__ __forceinline__ void async_load_b128(unsigned lds_off,
                                                const _Float16* gbase,
                                                int imm_bytes) {
    asm volatile("global_load_async_to_lds_b128 %0, %1, off offset:%2"
                 :: "v"(lds_off), "v"(gbase), "i"(imm_bytes) : "memory");
}
__device__ __forceinline__ void wait_async0() {
    asm volatile("s_wait_asynccnt 0x0" ::: "memory");
}
__device__ __forceinline__ void wait_async4() {
    asm volatile("s_wait_asynccnt 0x4" ::: "memory");
}

// ---------------------------------------------------------------------------
// Fused 25-step SNN scan. Block = 16 batch rows. mem1/spk1d in LDS, mem2 in
// VGPRs. GEMM2 per step: K=1024 in 32 slabs, fully unrolled, async-staged
// w2 slabs in a wave-private double buffer.
// ---------------------------------------------------------------------------
__global__ __launch_bounds__(256)
void snn_kernel(const float*    __restrict__ cur1,   // [4096][1024]
                const _Float16* __restrict__ w2h,    // [256][1024]
                const float*    __restrict__ b2,
                float*          __restrict__ out_spk,
                float*          __restrict__ out_mem)
{
    __shared__ float    smem1[16 * D_H];           // 64 KB
    __shared__ _Float16 sspk [16 * D_H];           // 32 KB
    __shared__ _Float16 wstage[8 * 2 * 1024];      // 32 KB: 8 waves x 2 x 2KB

    const int tid  = threadIdx.x;
    const int lane = tid & 31;
    const int wave = tid >> 5;
    const int m0   = blockIdx.x * 16;
    const int n0   = wave * 32;

    const int kofA = (lane >> 4) * 8;
    const int kofB = (lane >> 4) * 8;
    const int cb   = lane & 15;

    _Float16* wsp = &wstage[wave * 2048];          // parity stride 1024 halves
    const unsigned wsoff = (unsigned)(uintptr_t)wsp;

    // Loop-invariant staging addresses: issue "is" covers cols nl, chunk ch.
    // task = is*32+lane -> nl = is*8 + lane/4, ch = lane&3 (8 halves = 16B).
    const _Float16* gstage[4];
    unsigned loff0[4], loff1[4];
#pragma unroll
    for (int is = 0; is < 4; ++is) {
        const int nl = is * 8 + (lane >> 2);
        const int ch = lane & 3;
        gstage[is] = w2h + (size_t)(n0 + nl) * D_H + ch * 8;
        loff0[is]  = wsoff + (unsigned)(nl * 64 + ch * 16);
        loff1[is]  = loff0[is] + 2048u;
    }

#pragma unroll
    for (int i = 0; i < 64; ++i) smem1[i * 256 + tid] = 0.f;

    v8f zero = {0.f, 0.f, 0.f, 0.f, 0.f, 0.f, 0.f, 0.f};
    v8f mem2[2] = {zero, zero};
    float b2v[2];
#pragma unroll
    for (int t = 0; t < 2; ++t) b2v[t] = b2[n0 + t * 16 + cb];

    const _Float16* sA = &sspk[cb * D_H + kofA];   // A-frag base (this lane)
    const _Float16* sB0 = wsp + cb * 32 + kofB;    // B-frag tile-0 base
    // tile-1 base = sB0 + 512 halves; parity adds 1024 halves.

    __syncthreads();

    for (int step = 0; step < NSTEP; ++step) {
        // ---- layer-1 leaky integrate + spike + dropout ----
#pragma unroll 4
        for (int i = 0; i < 64; ++i) {
            int e   = i * 256 + tid;
            int row = e >> 10;
            int col = e & (D_H - 1);
            float m = smem1[e];
            float c = cur1[(size_t)(m0 + row) * D_H + col];
            float r1 = (m > THR) ? THR : 0.f;
            m = BETA * m + c - r1;
            smem1[e] = m;
            float s = (m > THR) ? 1.f : 0.f;
            unsigned u = (unsigned)((m0 + row) * D_H + col) +
                         0x9E3779B9u * (unsigned)(step + 1);
            u ^= u >> 16; u *= 0x7feb352du;
            u ^= u >> 15; u *= 0x846ca68bu;
            u ^= u >> 16;
            float sd = ((u & 0xFFFFFFu) < 13421773u) ? s * DROP_SCALE : 0.f;
            sspk[e] = (_Float16)sd;
        }
        __syncthreads();

        // ---- GEMM2: cur2 = spk1d @ w2^T + b2 (K=1024, 32 slabs) ----
        v8f acc[2];
#pragma unroll
        for (int t = 0; t < 2; ++t)
            acc[t] = (v8f){b2v[t], b2v[t], b2v[t], b2v[t],
                           b2v[t], b2v[t], b2v[t], b2v[t]};

        // stage slab 0 into parity 0
#pragma unroll
        for (int is = 0; is < 4; ++is)
            async_load_b128(loff0[is], gstage[is], 0);

        FragH a[2];
        a[0].q[0] = *(const uint4*)(sA);
        a[0].q[1] = *(const uint4*)(sA + 16);

#pragma unroll
        for (int s = 0; s < 32; ++s) {
            const int par = s & 1;
            if (s + 1 < 32) {                     // stage slab s+1 first...
#pragma unroll
                for (int is = 0; is < 4; ++is)
                    async_load_b128(par ? loff0[is] : loff1[is],
                                    gstage[is], (s + 1) * 64);
                wait_async4();  // ...in-order cnt<=4 => slab s resident
            } else {
                wait_async0();  // last slab: full drain
            }
            // B fragments for slab s (buffer-resident now)
            FragH b0, b1;
            {
                const _Float16* bp = sB0 + par * 1024;
                b0.q[0] = *(const uint4*)(bp);
                b0.q[1] = *(const uint4*)(bp + 16);
                b1.q[0] = *(const uint4*)(bp + 512);
                b1.q[1] = *(const uint4*)(bp + 528);
            }
            // A fragment for slab s+1 (sspk always resident; overlaps WMMAs)
            if (s + 1 < 32) {
                const _Float16* ap = sA + (s + 1) * 32;
                a[par ^ 1].q[0] = *(const uint4*)(ap);
                a[par ^ 1].q[1] = *(const uint4*)(ap + 16);
            }
            acc[0] = __builtin_amdgcn_wmma_f32_16x16x32_f16(
                false, a[par].v, false, b0.v, (short)0, acc[0], false, false);
            acc[1] = __builtin_amdgcn_wmma_f32_16x16x32_f16(
                false, a[par].v, false, b1.v, (short)0, acc[1], false, false);
        }

        // ---- layer-2 leaky integrate + outputs ----
        const int rbase = (lane >> 4) * 8;
        size_t outBase = (size_t)step * B_SZ * D_OUT;
#pragma unroll
        for (int t = 0; t < 2; ++t) {
            int col = n0 + t * 16 + cb;
#pragma unroll
            for (int j = 0; j < 8; ++j) {
                float m  = mem2[t][j];
                float r2 = (m > THR) ? THR : 0.f;
                m = BETA * m + acc[t][j] - r2;
                mem2[t][j] = m;
                float spk = (m > THR) ? 0.7310585786f : 0.5f;
                float sg  = 1.f / (1.f + __expf(-m));
                size_t o = outBase + (size_t)(m0 + rbase + j) * D_OUT + col;
                out_spk[o] = spk;
                out_mem[o] = sg;
            }
        }
        __syncthreads();   // sspk safe to overwrite next step
    }
}

// ---------------------------------------------------------------------------
__global__ void cvt_f32_f16(const float* __restrict__ src,
                            _Float16* __restrict__ dst, int n) {
    int i = blockIdx.x * blockDim.x + threadIdx.x;
    if (i < n) dst[i] = (_Float16)src[i];
}

// ---------------------------------------------------------------------------
// GEMM1: cur1[4096][1024] = x @ w1^T + b1. Wave = 16x64, K pipelined (x2).
// ---------------------------------------------------------------------------
__global__ __launch_bounds__(256)
void gemm1_kernel(const _Float16* __restrict__ xh,    // [4096][512]
                  const _Float16* __restrict__ w1h,   // [1024][512]
                  const float*    __restrict__ b1,
                  float*          __restrict__ cur1)  // [4096][1024]
{
    const int lane = threadIdx.x & 31;
    const int wave = threadIdx.x >> 5;
    const int m0   = blockIdx.x * 32 + (wave & 1) * 16;
    const int n0   = blockIdx.y * 256 + (wave >> 1) * 64;

    const int rowA = m0 + (lane & 15);
    const int kofA = (lane >> 4) * 8;
    const int kofB = (lane >> 4) * 8;
    const int cb   = lane & 15;

    v8f acc[4];
#pragma unroll
    for (int t = 0; t < 4; ++t) {
        float bv = b1[n0 + t * 16 + cb];
        acc[t] = (v8f){bv, bv, bv, bv, bv, bv, bv, bv};
    }

    const _Float16* apB = xh + (size_t)rowA * D_IN + kofA;

    FragH a[2], b[2][4];
    load_fragA(a[0], apB);
#pragma unroll
    for (int t = 0; t < 4; ++t)
        load_fragB(b[0][t], w1h + (size_t)(n0 + t * 16 + cb) * D_IN + kofB);

#pragma unroll 2
    for (int s = 0; s < 16; ++s) {              // 16 slabs of K=32
        const int cur = s & 1;
        if (s + 1 < 16) {                       // prefetch next slab
            const int kt = (s + 1) * 32;
            load_fragA(a[cur ^ 1], apB + kt);
#pragma unroll
            for (int t = 0; t < 4; ++t)
                load_fragB(b[cur ^ 1][t],
                           w1h + (size_t)(n0 + t * 16 + cb) * D_IN + kt + kofB);
        }
#pragma unroll
        for (int t = 0; t < 4; ++t)
            acc[t] = __builtin_amdgcn_wmma_f32_16x16x32_f16(
                false, a[cur].v, false, b[cur][t].v, (short)0, acc[t],
                false, false);
    }

    const int rbase = m0 + (lane >> 4) * 8;
#pragma unroll
    for (int t = 0; t < 4; ++t) {
        int col = n0 + t * 16 + cb;
#pragma unroll
        for (int j = 0; j < 8; ++j)
            cur1[(size_t)(rbase + j) * D_H + col] = acc[t][j];
    }
}

// ---------------------------------------------------------------------------
extern "C" void kernel_launch(void* const* d_in, const int* in_sizes, int n_in,
                              void* d_out, int out_size, void* d_ws,
                              size_t ws_size, hipStream_t stream) {
    const float* x  = (const float*)d_in[0];
    const float* w1 = (const float*)d_in[1];
    const float* b1 = (const float*)d_in[2];
    const float* w2 = (const float*)d_in[3];
    const float* b2 = (const float*)d_in[4];
    float* out = (float*)d_out;

    char* ws = (char*)d_ws;
    _Float16* xh   = (_Float16*)(ws);                     // 4 MB
    _Float16* w1h  = (_Float16*)(ws + ((size_t)4 << 20)); // 1 MB
    _Float16* w2h  = (_Float16*)(ws + ((size_t)5 << 20)); // 0.5 MB
    float*    cur1 = (float*)   (ws + ((size_t)6 << 20)); // 16 MB

    const int nx  = B_SZ * D_IN;
    const int nw1 = D_H * D_IN;
    const int nw2 = D_OUT * D_H;
    cvt_f32_f16<<<(nx  + 255) / 256, 256, 0, stream>>>(x,  xh,  nx);
    cvt_f32_f16<<<(nw1 + 255) / 256, 256, 0, stream>>>(w1, w1h, nw1);
    cvt_f32_f16<<<(nw2 + 255) / 256, 256, 0, stream>>>(w2, w2h, nw2);

    dim3 g1(B_SZ / 32, D_H / 256);
    gemm1_kernel<<<g1, 256, 0, stream>>>(xh, w1h, b1, cur1);

    float* out_spk = out;
    float* out_mem = out + (size_t)NSTEP * B_SZ * D_OUT;
    snn_kernel<<<B_SZ / 16, 256, 0, stream>>>(cur1, w2h, b2, out_spk, out_mem);
}
